// Attention_11519102287955
// MI455X (gfx1250) — matrix operand order repaired and checked by reading, not compiled, not run on hardware
//
#include <hip/hip_runtime.h>
#include <math.h>

typedef __attribute__((ext_vector_type(16))) _Float16 v16h;
typedef __attribute__((ext_vector_type(8)))  float    v8f;
typedef __attribute__((ext_vector_type(2)))  _Float16 h2;

#define DIM      768
#define HEADS    12
#define HEAD_DIM 64
#define SEQ      1024
#define BATCH    8
#define ROWS     8192      // BATCH*SEQ
#define QKV_N    2304

// ---- CDNA5 async global->LDS staging (ASYNCcnt path), guarded -------------
#if defined(__AMDGCN__) && defined(__has_builtin)
#if __has_builtin(__builtin_amdgcn_global_load_async_to_lds_b128)
#define USE_ASYNC_LDS 1
#endif
#endif
#ifndef USE_ASYNC_LDS
#define USE_ASYNC_LDS 0
#endif

#if USE_ASYNC_LDS
typedef int v4i __attribute__((vector_size(16)));
typedef __attribute__((address_space(1))) v4i* as1_v4i;
typedef __attribute__((address_space(3))) v4i* as3_v4i;
#endif

__device__ __forceinline__ void async_copy16(_Float16* dst, const _Float16* src) {
#if USE_ASYNC_LDS
  // generic->AS casts via integer round-trip; flat->local is a low-32-bit
  // truncation on AMDGPU.
  as1_v4i g = (as1_v4i)(unsigned long long)src;
  as3_v4i l = (as3_v4i)(unsigned int)(unsigned long long)dst;
  __builtin_amdgcn_global_load_async_to_lds_b128(g, l, 0, 0);
#else
  *(uint4*)dst = *(const uint4*)src;
#endif
}

__device__ __forceinline__ void async_join() {
#if USE_ASYNC_LDS
#if __has_builtin(__builtin_amdgcn_s_wait_asynccnt)
  __builtin_amdgcn_s_wait_asynccnt(0);
#else
  asm volatile("s_wait_asynccnt 0" ::: "memory");
#endif
#endif
}

// ---- WMMA fragment loaders (layouts per CDNA5 ISA 7.12.2) ----------------
// A (16x32 f16): lane m=lane&15; VGPR v holds K = (v>>2)*16 + half*8 + (v&3)*2
__device__ __forceinline__ v16h load_a_frag(const _Float16* base, int stride,
                                            int kbase, int lane) {
  const int m = lane & 15, hf = lane >> 4;
  const _Float16* rowp = base + m * stride + kbase;
  v16h a;
#pragma unroll
  for (int v = 0; v < 8; ++v) {
    int k = ((v >> 2) << 4) + (hf << 3) + ((v & 3) << 1);
    a[2 * v]     = rowp[k];
    a[2 * v + 1] = rowp[k + 1];
  }
  return a;
}

// B (32x16 f16): lane n=lane&15; VGPR v holds K = (v>>2)*16 + half*8 + (v&3)*2
// Element (k,n) lives at base[n*stride + k].
__device__ __forceinline__ v16h load_b_frag(const _Float16* base, int stride,
                                            int kbase, int lane) {
  const int n = lane & 15, hf = lane >> 4;
  const _Float16* colp = base + n * stride + kbase;
  v16h b;
#pragma unroll
  for (int v = 0; v < 8; ++v) {
    int k = ((v >> 2) << 4) + (hf << 3) + ((v & 3) << 1);
    b[2 * v]     = colp[k];
    b[2 * v + 1] = colp[k + 1];
  }
  return b;
}

// ---- one-shot converts ----------------------------------------------------
__global__ __launch_bounds__(256)
void cvt_f16(const float* __restrict__ in, _Float16* __restrict__ outp, int n) {
  int i = blockIdx.x * 256 + threadIdx.x;
  if (i < n) outp[i] = (_Float16)in[i];
}

// W (K x N, f32, row-major) -> Wt (N x K, f16, row-major)
__global__ __launch_bounds__(256)
void cvt_transpose(const float* __restrict__ W, _Float16* __restrict__ Wt,
                   int K, int N) {
  int nIdx = blockIdx.x * 32 + threadIdx.x;
  int kIdx = blockIdx.y * 8 + threadIdx.y;
  Wt[(size_t)nIdx * K + kIdx] = (_Float16)W[(size_t)kIdx * N + nIdx];
}

// ---- tiled GEMM: C = A(f16 MxK) @ Wt^T(f16 NxK) + bias --------------------
// 128x128 block tile, BK=32, 8 waves in 4(row)x2(col) grid, 8 wmma/wave/step
template <typename OutT>
__global__ __launch_bounds__(256)
void gemm_bias(const _Float16* __restrict__ A, const _Float16* __restrict__ Wt,
               const float* __restrict__ bias, OutT* __restrict__ C,
               int M, int N, int K) {
  __shared__ alignas(16) _Float16 sA[128 * 40];   // [m][k]
  __shared__ alignas(16) _Float16 sB[128 * 40];   // [n][k]
  const int t = threadIdx.x, lane = t & 31, w = t >> 5;
  const int r = w & 3, c = w >> 2;
  const int n0 = blockIdx.x * 128, m0 = blockIdx.y * 128;

  v8f zero = {};
  v8f acc[2][4];
#pragma unroll
  for (int s = 0; s < 2; ++s)
#pragma unroll
    for (int i = 0; i < 4; ++i) acc[s][i] = zero;

  for (int k0 = 0; k0 < K; k0 += 32) {
    __syncthreads();
#pragma unroll
    for (int i = 0; i < 2; ++i) {                  // A tile: 128 x 32
      int idx = i * 256 + t;
      int ml = idx >> 2, ck = (idx & 3) * 8;
      async_copy16(&sA[ml * 40 + ck], A + (size_t)(m0 + ml) * K + k0 + ck);
    }
#pragma unroll
    for (int i = 0; i < 2; ++i) {                  // B tile: 128 x 32
      int idx = i * 256 + t;
      int nl = idx >> 2, ck = (idx & 3) * 8;
      async_copy16(&sB[nl * 40 + ck], Wt + (size_t)(n0 + nl) * K + k0 + ck);
    }
    async_join();
    __syncthreads();
    v16h a0 = load_a_frag(sA + (r * 32) * 40, 40, 0, lane);
    v16h a1 = load_a_frag(sA + (r * 32 + 16) * 40, 40, 0, lane);
#pragma unroll
    for (int nt = 0; nt < 4; ++nt) {
      v16h b = load_b_frag(sB + (c * 64 + nt * 16) * 40, 40, 0, lane);
      acc[0][nt] = __builtin_amdgcn_wmma_f32_16x16x32_f16(
          false, a0, false, b, (short)0, acc[0][nt], false, false);
      acc[1][nt] = __builtin_amdgcn_wmma_f32_16x16x32_f16(
          false, a1, false, b, (short)0, acc[1][nt], false, false);
    }
  }

  const int hf = lane >> 4, n = lane & 15;
#pragma unroll
  for (int nt = 0; nt < 4; ++nt) {
    float bv = bias[n0 + c * 64 + nt * 16 + n];
#pragma unroll
    for (int s = 0; s < 2; ++s)
#pragma unroll
      for (int v = 0; v < 8; ++v) {
        int row = m0 + r * 32 + s * 16 + v + 8 * hf;
        C[(size_t)row * N + n0 + c * 64 + nt * 16 + n] = (OutT)(acc[s][nt][v] + bv);
      }
  }
}

// ---- Flash-style LSA attention --------------------------------------------
__global__ __launch_bounds__(256)
void lsa_attention(const _Float16* __restrict__ qkv,
                   const float* __restrict__ temperature,
                   _Float16* __restrict__ out) {
  __shared__ alignas(16) _Float16 sQ[128 * 72];     // [m][d]
  __shared__ alignas(16) _Float16 sK[64 * 72];      // [j][d]
  __shared__ alignas(16) _Float16 sVt[64 * 72];     // [d][j]
  __shared__ alignas(16) _Float16 sP[8 * 16 * 72];  // per-wave probs [m][j]

  const int t = threadIdx.x, lane = t & 31, w = t >> 5;
  const int rb = blockIdx.x, h = blockIdx.y, b = blockIdx.z;
  const float temp = temperature[h];
  const size_t rowBase = (size_t)b * SEQ;
  const int m0 = rb * 128;
  const int hf = lane >> 4, n = lane & 15;
  const h2* q2 = (const h2*)qkv;

  // stage Q tile (128 x 64): 1024 x 16B chunks, async
#pragma unroll
  for (int i = 0; i < 4; ++i) {
    int idx = i * 256 + t;
    int ml = idx >> 3, ck = (idx & 7) * 8;
    async_copy16(&sQ[ml * 72 + ck],
                 qkv + (rowBase + m0 + ml) * (size_t)QKV_N + h * HEAD_DIM + ck);
  }
  async_join();
  __syncthreads();

  v16h aq0 = load_a_frag(sQ + w * 16 * 72, 72, 0, lane);
  v16h aq1 = load_a_frag(sQ + w * 16 * 72, 72, 32, lane);

  float mrun[8], lrun[8];
  v8f zero = {};
  v8f acc[4];
#pragma unroll
  for (int i = 0; i < 8; ++i) { mrun[i] = -3.0e38f; lrun[i] = 0.f; }
#pragma unroll
  for (int i = 0; i < 4; ++i) acc[i] = zero;

  for (int kb = 0; kb < SEQ; kb += 64) {
    __syncthreads();
    // K tile (64 x 64): async row chunks
#pragma unroll
    for (int i = 0; i < 2; ++i) {
      int idx = i * 256 + t;
      int jl = idx >> 3, ck = (idx & 7) * 8;
      async_copy16(&sK[jl * 72 + ck],
                   qkv + (rowBase + kb + jl) * (size_t)QKV_N + DIM + h * HEAD_DIM + ck);
    }
    // V tile transposed (needs element shuffle -> regular ds stores)
#pragma unroll
    for (int i = 0; i < 8; ++i) {
      int idx = i * 256 + t;
      int jl = idx >> 5, d2 = (idx & 31) << 1;
      size_t gv = (rowBase + kb + jl) * (size_t)QKV_N + 2 * DIM + h * HEAD_DIM + d2;
      h2 vv = q2[gv >> 1];
      sVt[d2 * 72 + jl]       = vv[0];
      sVt[(d2 + 1) * 72 + jl] = vv[1];
    }
    async_join();
    __syncthreads();

    // S = (Q @ K^T) * temp, diag-masked
    float st[4][8];
#pragma unroll
    for (int ct = 0; ct < 4; ++ct) {
      v8f s = zero;
      v16h b0 = load_b_frag(sK + ct * 16 * 72, 72, 0, lane);
      s = __builtin_amdgcn_wmma_f32_16x16x32_f16(false, aq0, false, b0,
                                                 (short)0, s, false, false);
      v16h b1 = load_b_frag(sK + ct * 16 * 72, 72, 32, lane);
      s = __builtin_amdgcn_wmma_f32_16x16x32_f16(false, aq1, false, b1,
                                                 (short)0, s, false, false);
      int j = kb + ct * 16 + n;
#pragma unroll
      for (int v = 0; v < 8; ++v) {
        int irow = m0 + w * 16 + v + 8 * hf;
        float sv = s[v] * temp;
        st[ct][v] = (irow == j) ? -3.0e38f : sv;
      }
    }

    // online softmax (rows striped m = v + 8*half; reduce within 16-lane half)
#pragma unroll
    for (int v = 0; v < 8; ++v) {
      float rm = fmaxf(fmaxf(st[0][v], st[1][v]), fmaxf(st[2][v], st[3][v]));
#pragma unroll
      for (int off = 1; off < 16; off <<= 1)
        rm = fmaxf(rm, __shfl_xor(rm, off, 32));
      float mn = fmaxf(mrun[v], rm);
      float scale = __expf(mrun[v] - mn);
      float rsum = 0.f;
#pragma unroll
      for (int ct = 0; ct < 4; ++ct) {
        float p = __expf(st[ct][v] - mn);
        st[ct][v] = p;
        rsum += p;
      }
#pragma unroll
      for (int off = 1; off < 16; off <<= 1)
        rsum += __shfl_xor(rsum, off, 32);
      lrun[v] = lrun[v] * scale + rsum;
      mrun[v] = mn;
#pragma unroll
      for (int nt = 0; nt < 4; ++nt) acc[nt][v] = acc[nt][v] * scale;
      int m = v + 8 * hf;
#pragma unroll
      for (int ct = 0; ct < 4; ++ct)
        sP[w * 1152 + m * 72 + ct * 16 + n] = (_Float16)st[ct][v];
    }

    // O += P @ V
#pragma unroll
    for (int kc = 0; kc < 2; ++kc) {
      v16h ap = load_a_frag(sP + w * 1152, 72, kc * 32, lane);
#pragma unroll
      for (int nt = 0; nt < 4; ++nt) {
        v16h bv = load_b_frag(sVt + nt * 16 * 72, 72, kc * 32, lane);
        acc[nt] = __builtin_amdgcn_wmma_f32_16x16x32_f16(
            false, ap, false, bv, (short)0, acc[nt], false, false);
      }
    }
  }

  // normalize + write (B, N, H*D) f16 (proj GEMM consumes f16)
#pragma unroll
  for (int v = 0; v < 8; ++v) {
    float inv = 1.0f / lrun[v];
    int irow = m0 + w * 16 + v + 8 * hf;
    size_t obase = (rowBase + irow) * (size_t)DIM + h * HEAD_DIM;
#pragma unroll
    for (int nt = 0; nt < 4; ++nt)
      out[obase + nt * 16 + n] = (_Float16)(acc[nt][v] * inv);
  }
}

extern "C" void kernel_launch(void* const* d_in, const int* in_sizes, int n_in,
                              void* d_out, int out_size, void* d_ws, size_t ws_size,
                              hipStream_t stream) {
  (void)in_sizes; (void)n_in; (void)out_size; (void)ws_size;
  const float* x      = (const float*)d_in[0];
  const float* w_qkv  = (const float*)d_in[1];
  const float* b_qkv  = (const float*)d_in[2];
  const float* w_proj = (const float*)d_in[3];
  const float* b_proj = (const float*)d_in[4];
  const float* temp   = (const float*)d_in[5];

  auto align256 = [](size_t v) { return (v + 255) & ~(size_t)255; };
  char* ws = (char*)d_ws;
  _Float16* qkv    = (_Float16*)ws;  ws += align256((size_t)ROWS * QKV_N * 2);
  _Float16* attnH  = (_Float16*)ws;  ws += align256((size_t)ROWS * DIM * 2);
  _Float16* xh     = (_Float16*)ws;  ws += align256((size_t)ROWS * DIM * 2);
  _Float16* wqkvT  = (_Float16*)ws;  ws += align256((size_t)DIM * QKV_N * 2);
  _Float16* wprojT = (_Float16*)ws;  ws += align256((size_t)DIM * DIM * 2);

  const int nX = ROWS * DIM;
  cvt_f16<<<(nX + 255) / 256, 256, 0, stream>>>(x, xh, nX);
  cvt_transpose<<<dim3(QKV_N / 32, DIM / 8), dim3(32, 8), 0, stream>>>(
      w_qkv, wqkvT, DIM, QKV_N);
  cvt_transpose<<<dim3(DIM / 32, DIM / 8), dim3(32, 8), 0, stream>>>(
      w_proj, wprojT, DIM, DIM);

  gemm_bias<_Float16><<<dim3(QKV_N / 128, ROWS / 128), 256, 0, stream>>>(
      xh, wqkvT, b_qkv, qkv, ROWS, QKV_N, DIM);
  lsa_attention<<<dim3(SEQ / 128, HEADS, BATCH), 256, 0, stream>>>(
      qkv, temp, attnH);
  gemm_bias<float><<<dim3(DIM / 128, ROWS / 128), 256, 0, stream>>>(
      attnH, wprojT, b_proj, (float*)d_out, ROWS, DIM, DIM);
}
